// DecoderLSTM_73993696575788
// MI455X (gfx1250) — matrix operand order repaired and checked by reading, not compiled, not run on hardware
//
#include <hip/hip_runtime.h>

// Problem dims (match reference)
#define V_  32000
#define E_  512
#define H_  1024
#define B_  64
#define T_  32
#define G4H 4096   // 4*H

typedef __attribute__((ext_vector_type(16))) _Float16 v16h;
typedef __attribute__((ext_vector_type(8)))  float    v8f;

union U16H { uint4 u[2]; v16h h; };

// ---------------------------------------------------------------------------
// Packed fragment layouts (wave32, V_WMMA_F32_16X16X32_F16):
// Each (tile, kchunk) is 512 halves = 1KB, stored so lane L reads its 16
// halves with two coalesced 16B loads: halves [L*8 .. L*8+7] and
// [256 + L*8 .. +7].
//
// A (16x32, MxK)  lane = ((k&31)>>3 &1)<<4 | (m&15),  half i = ((k>>4)&1)<<3 | (k&7)
// B (32x16, KxN)  lane = ((k&31)>>3 &1)<<4 | (n&15),  half i = ((k>>4)&1)<<3 | (k&7)
// C/D (16x16 f32) lane l, vgpr r -> M = (l>>4)*8 + r, N = l&15
// ---------------------------------------------------------------------------

__device__ __forceinline__ v16h load_chunk(const _Float16* base, int lane) {
    const uint4* p = (const uint4*)base;
    U16H w;
    w.u[0] = p[lane];        // halves lane*8 .. +7
    w.u[1] = p[32 + lane];   // halves 256 + lane*8 .. +7
    return w.h;
}

__device__ __forceinline__ size_t a_pack_index(int m, int k, int KC) {
    int mtile = m >> 4, r = m & 15;
    int kc = k >> 5, kk = k & 31;
    int a  = (kk >> 4) & 1;           // upper-16 K half -> half-index MSB
    int b2 = (kk >> 3) & 1;           // selects lane group
    int c  = kk & 7;
    int lane = (b2 << 4) | r;
    return ((size_t)(mtile * KC + kc) << 9) + ((size_t)a << 8) + ((size_t)lane << 3) + c;
}

// One-time fp32 -> f16 weight repack into B-fragment layout. src is [K,N] row-major.
__global__ void pack_b_kernel(const float* __restrict__ src, _Float16* __restrict__ dst,
                              int K, int N) {
    size_t id = (size_t)blockIdx.x * blockDim.x + threadIdx.x;
    if (id >= (size_t)K * N) return;
    int n = (int)(id % N);
    int k = (int)(id / N);
    int ntile = n >> 4, nin = n & 15;
    int kc = k >> 5,  kk = k & 31;
    int lane = (((kk >> 3) & 1) << 4) | nin;
    int i = (((kk >> 4) & 1) << 3) | (kk & 7);
    size_t dsti = ((size_t)(ntile * (K >> 5) + kc) << 9) + ((size_t)(i >> 3) << 8)
                + ((size_t)lane << 3) + (i & 7);
    dst[dsti] = (_Float16)src[id];
}

// Initialize per-call state: h/c fp32 state, packed-A h fragments, first tokens.
__global__ void init_state_kernel(const float* __restrict__ h0, const float* __restrict__ c0,
                                  const int* __restrict__ sos, float* __restrict__ hst,
                                  float* __restrict__ cst, int* __restrict__ tok,
                                  _Float16* __restrict__ Ah) {
    int id = blockIdx.x * blockDim.x + threadIdx.x;   // B*H
    int b = id >> 10, k = id & (H_ - 1);
    hst[id] = h0[id];
    cst[id] = c0[id];
    Ah[a_pack_index(b, k, H_ / 32)] = (_Float16)h0[id];
    if (id < B_) tok[id] = sos[id];
}

// Per-step embedding gather -> packed A fragments (f16).
__global__ void gather_embed_kernel(const float* __restrict__ emb, const int* __restrict__ tok,
                                    _Float16* __restrict__ Aemb) {
    int id = blockIdx.x * blockDim.x + threadIdx.x;   // B*E
    int b = id >> 9, e = id & (E_ - 1);
    float v = emb[(size_t)tok[b] * E_ + e];
    Aemb[a_pack_index(b, e, E_ / 32)] = (_Float16)v;
}

// Software-pipelined fragment-GEMM core: acc[0..3] += A(mtile,:) * B(n0..n0+3,:)
// over KC k-chunks. Prefetches chunk kc+1 (A + 4 B frags) while chunk kc's
// 4 WMMAs execute, so loads stay in flight behind the matrix pipe.
__device__ __forceinline__ void gemm_frag_pipelined(
        const _Float16* __restrict__ Abase,   // + (mtile*KC)<<9 already applied
        const _Float16* __restrict__ Bbase,   // weight fragments
        int n0, int KC, int lane, v8f acc[4]) {
    const _Float16* b0p = Bbase + ((size_t)((n0 + 0) * KC) << 9);
    const _Float16* b1p = Bbase + ((size_t)((n0 + 1) * KC) << 9);
    const _Float16* b2p = Bbase + ((size_t)((n0 + 2) * KC) << 9);
    const _Float16* b3p = Bbase + ((size_t)((n0 + 3) * KC) << 9);

    v16h a  = load_chunk(Abase, lane);
    v16h b0 = load_chunk(b0p, lane);
    v16h b1 = load_chunk(b1p, lane);
    v16h b2 = load_chunk(b2p, lane);
    v16h b3 = load_chunk(b3p, lane);

    for (int kc = 0; kc < KC; ++kc) {
        // Prefetch next chunk (last iteration redundantly reloads current: harmless).
        int kn = (kc + 1 < KC) ? (kc + 1) : kc;
        size_t off = (size_t)kn << 9;
        v16h an  = load_chunk(Abase + off, lane);
        v16h c0  = load_chunk(b0p + off, lane);
        v16h c1  = load_chunk(b1p + off, lane);
        v16h c2  = load_chunk(b2p + off, lane);
        v16h c3  = load_chunk(b3p + off, lane);

        acc[0] = __builtin_amdgcn_wmma_f32_16x16x32_f16(false, a, false, b0,
                                                        (short)0, acc[0], false, false);
        acc[1] = __builtin_amdgcn_wmma_f32_16x16x32_f16(false, a, false, b1,
                                                        (short)0, acc[1], false, false);
        acc[2] = __builtin_amdgcn_wmma_f32_16x16x32_f16(false, a, false, b2,
                                                        (short)0, acc[2], false, false);
        acc[3] = __builtin_amdgcn_wmma_f32_16x16x32_f16(false, a, false, b3,
                                                        (short)0, acc[3], false, false);

        a = an; b0 = c0; b1 = c1; b2 = c2; b3 = c3;
    }
}

// gates[64,4096] = emb@Wx + h@Wh + bx + bh  (f16 WMMA, f32 accum)
// 256 waves: 4 M-tiles x 64 N-groups (4 N-tiles each).
__global__ void __launch_bounds__(256)
gates_gemm_kernel(const _Float16* __restrict__ Aemb, const _Float16* __restrict__ Wxp,
                  const _Float16* __restrict__ Ah,   const _Float16* __restrict__ Whp,
                  const float* __restrict__ bx, const float* __restrict__ bh,
                  float* __restrict__ gates) {
    int lane = threadIdx.x & 31;
    int wg = blockIdx.x * (blockDim.x >> 5) + (threadIdx.x >> 5);
    int mtile = wg >> 6;
    int n0 = (wg & 63) * 4;
    v8f acc[4] = {};
    const int KC1 = E_ / 32;   // 16
    gemm_frag_pipelined(Aemb + ((size_t)(mtile * KC1) << 9), Wxp, n0, KC1, lane, acc);
    const int KC2 = H_ / 32;   // 32
    gemm_frag_pipelined(Ah + ((size_t)(mtile * KC2) << 9), Whp, n0, KC2, lane, acc);

    int ncol = lane & 15;
    int mbase = mtile * 16 + (lane >> 4) * 8;
#pragma unroll
    for (int j = 0; j < 4; ++j) {
        int n = (n0 + j) * 16 + ncol;
        float bias = bx[n] + bh[n];
#pragma unroll
        for (int r = 0; r < 8; ++r)
            gates[(size_t)(mbase + r) * G4H + n] = acc[j][r] + bias;
    }
}

__device__ __forceinline__ float sigmoidf_(float x) { return 1.0f / (1.0f + __expf(-x)); }

// LSTM pointwise: c' = sig(f)*c + sig(i)*tanh(g); h' = sig(o)*tanh(c'). Repacks h' -> A frags.
__global__ void lstm_pointwise_kernel(const float* __restrict__ gates, float* __restrict__ cst,
                                      float* __restrict__ hst, _Float16* __restrict__ Ah) {
    int id = blockIdx.x * blockDim.x + threadIdx.x;   // B*H
    int b = id >> 10, k = id & (H_ - 1);
    const float* g = gates + (size_t)b * G4H;
    float f  = sigmoidf_(g[k]);
    float i  = sigmoidf_(g[H_ + k]);
    float gg = tanhf(g[2 * H_ + k]);
    float o  = sigmoidf_(g[3 * H_ + k]);
    float c = f * cst[id] + i * gg;
    float h = o * tanhf(c);
    cst[id] = c;
    hst[id] = h;
    Ah[a_pack_index(b, k, H_ / 32)] = (_Float16)h;
}

// logits[64,32000] = h@Wout + bout. 2000 waves: 4 M-tiles x 500 N-groups (4 N-tiles each).
__global__ void __launch_bounds__(256)
logits_gemm_kernel(const _Float16* __restrict__ Ah, const _Float16* __restrict__ Wop,
                   const float* __restrict__ bout, float* __restrict__ out) {
    int lane = threadIdx.x & 31;
    int wg = blockIdx.x * (blockDim.x >> 5) + (threadIdx.x >> 5);   // 0..1999
    int mtile = wg / 500;
    int n0 = (wg % 500) * 4;
    const int KC = H_ / 32;    // 32
    v8f acc[4] = {};
    gemm_frag_pipelined(Ah + ((size_t)(mtile * KC) << 9), Wop, n0, KC, lane, acc);

    int ncol = lane & 15;
    int mbase = mtile * 16 + (lane >> 4) * 8;
#pragma unroll
    for (int j = 0; j < 4; ++j) {
        int n = (n0 + j) * 16 + ncol;
        float bias = bout[n];
#pragma unroll
        for (int r = 0; r < 8; ++r)
            out[(size_t)(mbase + r) * V_ + n] = acc[j][r] + bias;
    }
}

// Greedy top-1 per row (first-index tie-break like jnp.argmax). 1 block per batch row.
__global__ void argmax_kernel(const float* __restrict__ logits, int* __restrict__ tok) {
    __shared__ float sv[256];
    __shared__ int   si[256];
    int b = blockIdx.x;
    const float* row = logits + (size_t)b * V_;
    float best = -3.4e38f;
    int bi = 0;
    for (int n = threadIdx.x; n < V_; n += 256) {
        float v = row[n];
        if (v > best) { best = v; bi = n; }
    }
    sv[threadIdx.x] = best;
    si[threadIdx.x] = bi;
    __syncthreads();
    for (int s = 128; s > 0; s >>= 1) {
        if ((int)threadIdx.x < s) {
            float ov = sv[threadIdx.x + s];
            int   oi = si[threadIdx.x + s];
            if (ov > sv[threadIdx.x] || (ov == sv[threadIdx.x] && oi < si[threadIdx.x])) {
                sv[threadIdx.x] = ov;
                si[threadIdx.x] = oi;
            }
        }
        __syncthreads();
    }
    if (threadIdx.x == 0) tok[b] = si[0];
}

// ---------------------------------------------------------------------------
extern "C" void kernel_launch(void* const* d_in, const int* in_sizes, int n_in,
                              void* d_out, int out_size, void* d_ws, size_t ws_size,
                              hipStream_t stream) {
    const float* embedding = (const float*)d_in[0];
    const float* Wx   = (const float*)d_in[1];
    const float* bx   = (const float*)d_in[2];
    const float* Wh   = (const float*)d_in[3];
    const float* bh   = (const float*)d_in[4];
    const float* Wout = (const float*)d_in[5];
    const float* bout = (const float*)d_in[6];
    const float* h0   = (const float*)d_in[7];
    const float* c0   = (const float*)d_in[8];
    const int*   sos  = (const int*)d_in[9];
    float* out = (float*)d_out;

    // Workspace layout (all offsets 256B-aligned), ~79.9 MB total.
    char* ws = (char*)d_ws;
    _Float16* Wxp  = (_Float16*)(ws + 0);          //  512*4096 f16 =  4,194,304 B
    _Float16* Whp  = (_Float16*)(ws + 4194304);    // 1024*4096 f16 =  8,388,608 B
    _Float16* Wop  = (_Float16*)(ws + 12582912);   // 1024*32000 f16 = 65,536,000 B
    _Float16* Aemb = (_Float16*)(ws + 78118912);   //   64*512 f16
    _Float16* Ah   = (_Float16*)(ws + 78184448);   //  64*1024 f16
    float*    gates= (float*)   (ws + 78315520);   //  64*4096 f32
    float*    hst  = (float*)   (ws + 79364096);   //  64*1024 f32
    float*    cst  = (float*)   (ws + 79626240);   //  64*1024 f32
    int*      tok  = (int*)     (ws + 79888384);   //  64 int

    // One-time per call: repack weights to f16 WMMA fragments (L2-resident, ~76MB).
    pack_b_kernel<<<(E_ * G4H) / 256, 256, 0, stream>>>(Wx, Wxp, E_, G4H);
    pack_b_kernel<<<(H_ * G4H) / 256, 256, 0, stream>>>(Wh, Whp, H_, G4H);
    pack_b_kernel<<<(H_ * V_) / 256, 256, 0, stream>>>(Wout, Wop, H_, V_);
    init_state_kernel<<<(B_ * H_) / 256, 256, 0, stream>>>(h0, c0, sos, hst, cst, tok, Ah);

    for (int t = 0; t < T_; ++t) {
        float* out_t = out + (size_t)t * B_ * V_;
        gather_embed_kernel<<<(B_ * E_) / 256, 256, 0, stream>>>(embedding, tok, Aemb);
        gates_gemm_kernel<<<32, 256, 0, stream>>>(Aemb, Wxp, Ah, Whp, bx, bh, gates);
        lstm_pointwise_kernel<<<(B_ * H_) / 256, 256, 0, stream>>>(gates, cst, hst, Ah);
        logits_gemm_kernel<<<250, 256, 0, stream>>>(Ah, Wop, bout, out_t);
        argmax_kernel<<<B_, 256, 0, stream>>>(out_t, tok);
    }
}
